// RWKV7TimeMixing_56642028699977
// MI455X (gfx1250) — matrix operand order repaired and checked by reading, not compiled, not run on hardware
//
#include <hip/hip_runtime.h>
#include <hip/hip_bf16.h>
#include <math.h>

#define BB 8
#define TT 2048
#define CC 1024
#define HH 16
#define DDIM 64
#define MM (BB*TT)   // 16384 rows

typedef __attribute__((ext_vector_type(16))) __bf16 v16bf;
typedef __attribute__((ext_vector_type(8)))  float  v8f;
typedef __attribute__((ext_vector_type(8)))  unsigned short v8us;

union BF16Frag { unsigned short u[16]; v16bf v; };

__device__ __forceinline__ unsigned short f2bf(float f) {
  unsigned int x = __float_as_uint(f);
  unsigned int r = x + 0x7FFFu + ((x >> 16) & 1u);   // round to nearest even
  return (unsigned short)(r >> 16);
}

__device__ __forceinline__ float sigmoidf_(float x) { return 1.0f / (1.0f + __expf(-x)); }

// ---------------------------------------------------------------- weights -> bf16
__global__ void k_conv_weights(const float* __restrict__ w0, const float* __restrict__ w1,
                               const float* __restrict__ w2, const float* __restrict__ w3,
                               const float* __restrict__ w4, unsigned short* __restrict__ wb) {
  size_t idx = (size_t)blockIdx.x * blockDim.x + threadIdx.x;
  const size_t n = (size_t)5 * CC * CC;
  if (idx >= n) return;
  int slot = (int)(idx / ((size_t)CC * CC));
  size_t off = idx % ((size_t)CC * CC);
  const float* src = slot == 0 ? w0 : slot == 1 ? w1 : slot == 2 ? w2 : slot == 3 ? w3 : w4;
  wb[idx] = f2bf(src[off]);
}

// ------------------------------------------------- time-shift mix -> bf16 (4 proj)
__global__ void k_mix(const float* __restrict__ x, const float* __restrict__ tr,
                      const float* __restrict__ tk, const float* __restrict__ tv,
                      const float* __restrict__ tg, unsigned short* __restrict__ xm) {
  size_t idx = (size_t)blockIdx.x * blockDim.x + threadIdx.x;
  const size_t n = (size_t)MM * CC;
  if (idx >= n) return;
  size_t m = idx / CC; int c = (int)(idx % CC);
  int t = (int)(m % TT);
  float xc = x[idx];
  float xs = (t == 0) ? 0.f : x[idx - CC];
  float wr = tr[c], wk = tk[c], wv = tv[c], wg = tg[c];
  xm[idx]           = f2bf(xc * wr + xs * (1.f - wr));
  xm[idx + n]       = f2bf(xc * wk + xs * (1.f - wk));
  xm[idx + 2 * n]   = f2bf(xc * wv + xs * (1.f - wv));
  xm[idx + 3 * n]   = f2bf(xc * wg + xs * (1.f - wg));
}

// ------------------------------------------------- fused r/k/v/g GEMM (bf16 WMMA)
// Y = X @ W^T + b, activation per projection.
// Register-blocked: each wave computes 2 m-tiles x 4 n-tiles (32x64 output),
// 8 WMMAs per k-step from 12 b128 loads.
__global__ void k_gemm_qkvg(const unsigned short* __restrict__ xm, const unsigned short* __restrict__ wb,
                            const float* __restrict__ br, const float* __restrict__ bk,
                            const float* __restrict__ bv, const float* __restrict__ bg,
                            float* __restrict__ rbuf, float* __restrict__ kbuf,
                            float* __restrict__ vbuf, float* __restrict__ gbuf) {
  const int lane  = threadIdx.x & 31;
  const int wave  = threadIdx.x >> 5;
  const int m0    = blockIdx.x * 32;          // 2 m-tiles
  const int nst   = blockIdx.y * 4 + wave;    // n-supertile of 64 columns
  const int p     = blockIdx.z;
  const unsigned short* A  = xm + (size_t)p * MM * CC;
  const unsigned short* Wt = wb + (size_t)p * CC * CC;
  const int l15  = lane & 15;
  const int kbA  = (lane < 16) ? 0 : 8;       // A-frag K sub-block per ISA layout
  const int kbB  = (lane < 16) ? 0 : 8;       // B-frag K sub-block per ISA layout
  const unsigned short* arow0 = A + (size_t)(m0 + l15) * CC + kbA;
  const unsigned short* arow1 = arow0 + (size_t)16 * CC;
  const unsigned short* brow[4];
#pragma unroll
  for (int j = 0; j < 4; ++j)
    brow[j] = Wt + (size_t)(nst * 64 + j * 16 + l15) * CC + kbB;
  const v8f zero = {0.f, 0.f, 0.f, 0.f, 0.f, 0.f, 0.f, 0.f};
  v8f acc[2][4];
#pragma unroll
  for (int mi = 0; mi < 2; ++mi)
#pragma unroll
    for (int nj = 0; nj < 4; ++nj) acc[mi][nj] = zero;
  for (int k0 = 0; k0 < CC; k0 += 32) {
    __builtin_prefetch(arow0 + k0 + 128, 0, 1);
    __builtin_prefetch(arow1 + k0 + 128, 0, 1);
    BF16Frag af[2], bfr[4];
    {
      v8us lo0 = *(const v8us*)(arow0 + k0);
      v8us hi0 = *(const v8us*)(arow0 + k0 + 16);
      v8us lo1 = *(const v8us*)(arow1 + k0);
      v8us hi1 = *(const v8us*)(arow1 + k0 + 16);
#pragma unroll
      for (int e = 0; e < 8; ++e) {
        af[0].u[e] = lo0[e]; af[0].u[8 + e] = hi0[e];
        af[1].u[e] = lo1[e]; af[1].u[8 + e] = hi1[e];
      }
    }
#pragma unroll
    for (int nj = 0; nj < 4; ++nj) {
      v8us lo = *(const v8us*)(brow[nj] + k0);
      v8us hi = *(const v8us*)(brow[nj] + k0 + 16);
#pragma unroll
      for (int e = 0; e < 8; ++e) { bfr[nj].u[e] = lo[e]; bfr[nj].u[8 + e] = hi[e]; }
    }
#pragma unroll
    for (int mi = 0; mi < 2; ++mi)
#pragma unroll
      for (int nj = 0; nj < 4; ++nj)
        acc[mi][nj] = __builtin_amdgcn_wmma_f32_16x16x32_bf16(
            false, af[mi].v, false, bfr[nj].v, (short)0, acc[mi][nj], false, false);
  }
  const float* bias = (p == 0) ? br : (p == 1) ? bk : (p == 2) ? bv : bg;
  float* out = (p == 0) ? rbuf : (p == 1) ? kbuf : (p == 2) ? vbuf : gbuf;
  int mbase = m0 + ((lane < 16) ? 0 : 8);
#pragma unroll
  for (int nj = 0; nj < 4; ++nj) {
    int ncol = nst * 64 + nj * 16 + l15;
    float bn = bias[ncol];
#pragma unroll
    for (int mi = 0; mi < 2; ++mi) {
#pragma unroll
      for (int rr = 0; rr < 8; ++rr) {
        float y = acc[mi][nj][rr] + bn;
        if (p == 0) y = sigmoidf_(y);              // r: sigmoid
        else if (p == 3) y = y * sigmoidf_(y);     // g: silu
        out[(size_t)(mbase + mi * 16 + rr) * CC + ncol] = y;
      }
    }
  }
}

// ------------------------------------------------- per-head l2norm over D=64
__global__ void k_l2norm64(float* __restrict__ buf) {
  const int lane = threadIdx.x & 31;
  const int wave = threadIdx.x >> 5;
  size_t g = (size_t)blockIdx.x * 8 + wave;
  float* p = buf + g * 64;
  float a = p[lane], c = p[lane + 32];
  float ss = a * a + c * c;
#pragma unroll
  for (int m = 16; m >= 1; m >>= 1) ss += __shfl_xor(ss, m, 32);
  float inv = 1.0f / fmaxf(sqrtf(ss), 1e-12f);
  p[lane] = a * inv; p[lane + 32] = c * inv;
}

// ------------------------------------------------- sequential delta-rule scan
// one block per (b,h); 64x64 fp32 state in registers, 16 elems/thread.
// r/k/v for step t+1 are double-buffered into LDS with async-to-LDS loads
// (ASYNCcnt) so the fetch overlaps the whole step-t computation.
__global__ void k_scan(const float* __restrict__ rbuf, const float* __restrict__ kbuf,
                       const float* __restrict__ vbuf, const float* __restrict__ decay,
                       const float* __restrict__ eta, float* __restrict__ wkv,
                       float* __restrict__ state_out) {
  __shared__ float stage[2][192];   // [k(64) | v(64) | r(64)] double-buffered
  __shared__ float lerr[64], wred[8];
  const int tid  = threadIdx.x;
  const int lane = tid & 31;
  const int wave = tid >> 5;
  const int bh = blockIdx.x;
  const int b = bh >> 4, h = bh & 15;
  const float dec = sigmoidf_(decay[h]);
  const float et  = sigmoidf_(eta[h]);
  const int i  = tid >> 2;       // state row 0..63
  const int q  = tid & 3;        // column quarter
  const int j0 = q * 16;
  float s[16];
#pragma unroll
  for (int e = 0; e < 16; ++e) s[e] = 0.f;
  const size_t cbase = ((size_t)b * TT) * CC + (size_t)h * 64;
  const float* srcp = (tid < 64) ? kbuf : (tid < 128) ? vbuf : rbuf;
  const int d = tid & 63;
  const unsigned ldsoff0 = (unsigned)(size_t)(void*)&stage[0][tid];
  const unsigned ldsoff1 = (unsigned)(size_t)(void*)&stage[1][tid];
  // prologue: async-load t=0 into stage[0]
  if (tid < 192) {
    unsigned long long g0 = (unsigned long long)(const void*)(srcp + cbase + d);
    asm volatile("global_load_async_to_lds_b32 %0, %1, off" :: "v"(ldsoff0), "v"(g0) : "memory");
  }
  for (int t = 0; t < TT; ++t) {
    const int cur = t & 1;
    const size_t base = cbase + (size_t)t * CC;
    asm volatile("s_wait_asynccnt 0x0" ::: "memory");
    __syncthreads();                       // stage[cur] ready for all waves
    if (tid < 192 && (t + 1) < TT) {       // overlap next step's fetch
      unsigned long long gn = (unsigned long long)(const void*)(srcp + base + CC + d);
      unsigned dst = cur ? ldsoff0 : ldsoff1;
      asm volatile("global_load_async_to_lds_b32 %0, %1, off" :: "v"(dst), "v"(gn) : "memory");
    }
    const float* lk = &stage[cur][0];
    const float* lv = &stage[cur][64];
    const float* lr = &stage[cur][128];
    // pred = S @ k  (4 lanes per row -> xor reduce within wave)
    float pp = 0.f;
#pragma unroll
    for (int e = 0; e < 16; ++e) pp += s[e] * lk[j0 + e];
    pp += __shfl_xor(pp, 1, 32);
    pp += __shfl_xor(pp, 2, 32);
    if (q == 0) lerr[i] = pp - lv[i];
    __syncthreads();
    float ei = lerr[i];
    // state = dec*S - et*err(x)k ; accumulate Frobenius norm^2
    float nsq = 0.f;
#pragma unroll
    for (int e = 0; e < 16; ++e) {
      float sv = dec * s[e] - et * ei * lk[j0 + e];
      s[e] = sv;
      nsq += sv * sv;
    }
#pragma unroll
    for (int m = 16; m >= 1; m >>= 1) nsq += __shfl_xor(nsq, m, 32);
    if (lane == 0) wred[wave] = nsq;
    __syncthreads();
    float tot = wred[0] + wred[1] + wred[2] + wred[3] + wred[4] + wred[5] + wred[6] + wred[7];
    float nrm = fmaxf(sqrtf(tot), 1e-8f);
    float scale = (nrm > 16.0f) ? (16.0f / nrm) : 1.0f;   // max_norm = sqrt(64)*2
    float wv = 0.f;
#pragma unroll
    for (int e = 0; e < 16; ++e) {
      s[e] *= scale;
      wv += s[e] * lr[j0 + e];
    }
    wv += __shfl_xor(wv, 1, 32);
    wv += __shfl_xor(wv, 2, 32);
    if (q == 0) wkv[base + i] = wv;
    // loop-top barrier of next iteration protects stage/lerr/wred reuse
  }
  float* so = state_out + (size_t)bh * DDIM * DDIM;
#pragma unroll
  for (int e = 0; e < 16; ++e) so[i * DDIM + j0 + e] = s[e];
}

// ------------------------------------------------- groupnorm stats per (b,h)
__global__ void k_gnstats(const float* __restrict__ wkv, float* __restrict__ stats) {
  __shared__ float sred[8], ssred[8];
  const int tid = threadIdx.x, lane = tid & 31, wave = tid >> 5;
  const int bh = blockIdx.x; const int b = bh >> 4, h = bh & 15;
  const float* base = wkv + ((size_t)b * TT) * CC + (size_t)h * 64;
  float s = 0.f, ss = 0.f;
  for (int idx = tid; idx < TT * 64; idx += 256) {
    int t = idx >> 6, d = idx & 63;
    float v = base[(size_t)t * CC + d];
    s += v; ss += v * v;
  }
#pragma unroll
  for (int m = 16; m >= 1; m >>= 1) { s += __shfl_xor(s, m, 32); ss += __shfl_xor(ss, m, 32); }
  if (lane == 0) { sred[wave] = s; ssred[wave] = ss; }
  __syncthreads();
  if (tid == 0) {
    float S = 0.f, SS = 0.f;
    for (int w = 0; w < 8; ++w) { S += sred[w]; SS += ssred[w]; }
    float mean = S / (float)(TT * 64);
    float var  = SS / (float)(TT * 64) - mean * mean;
    stats[bh * 2]     = mean;
    stats[bh * 2 + 1] = rsqrtf(fmaxf(var, 0.f) + 1e-5f);
  }
}

// ------------------------------------------------- normalize + affine -> bf16
__global__ void k_gnnorm(const float* __restrict__ wkv, const float* __restrict__ stats,
                         const float* __restrict__ gw, const float* __restrict__ gb,
                         unsigned short* __restrict__ wkvn) {
  size_t idx = (size_t)blockIdx.x * blockDim.x + threadIdx.x;
  if (idx >= (size_t)MM * CC) return;
  size_t m = idx / CC; int c = (int)(idx % CC);
  int b = (int)(m / TT); int h = c >> 6;
  int bh = b * HH + h;
  float mean = stats[bh * 2], inv = stats[bh * 2 + 1];
  float y = (wkv[idx] - mean) * inv * gw[c] + gb[c];
  wkvn[idx] = f2bf(y);
}

// ------------------------------------------------- last_x copy
__global__ void k_lastx(const float* __restrict__ x, float* __restrict__ dst) {
  int idx = blockIdx.x * blockDim.x + threadIdx.x;
  if (idx >= BB * CC) return;
  int b = idx / CC, c = idx % CC;
  dst[idx] = x[((size_t)b * TT + (TT - 1)) * CC + c];
}

// ------------------------------------------------- output GEMM (bf16 WMMA) * g
// Same 2x4 register blocking as k_gemm_qkvg.
__global__ void k_gemm_out(const unsigned short* __restrict__ an, const unsigned short* __restrict__ wo,
                           const float* __restrict__ bo, const float* __restrict__ gbuf,
                           float* __restrict__ out) {
  const int lane  = threadIdx.x & 31;
  const int wave  = threadIdx.x >> 5;
  const int m0    = blockIdx.x * 32;
  const int nst   = blockIdx.y * 4 + wave;
  const int l15  = lane & 15;
  const int kbA  = (lane < 16) ? 0 : 8;
  const int kbB  = (lane < 16) ? 0 : 8;
  const unsigned short* arow0 = an + (size_t)(m0 + l15) * CC + kbA;
  const unsigned short* arow1 = arow0 + (size_t)16 * CC;
  const unsigned short* brow[4];
#pragma unroll
  for (int j = 0; j < 4; ++j)
    brow[j] = wo + (size_t)(nst * 64 + j * 16 + l15) * CC + kbB;
  const v8f zero = {0.f, 0.f, 0.f, 0.f, 0.f, 0.f, 0.f, 0.f};
  v8f acc[2][4];
#pragma unroll
  for (int mi = 0; mi < 2; ++mi)
#pragma unroll
    for (int nj = 0; nj < 4; ++nj) acc[mi][nj] = zero;
  for (int k0 = 0; k0 < CC; k0 += 32) {
    __builtin_prefetch(arow0 + k0 + 128, 0, 1);
    __builtin_prefetch(arow1 + k0 + 128, 0, 1);
    BF16Frag af[2], bfr[4];
    {
      v8us lo0 = *(const v8us*)(arow0 + k0);
      v8us hi0 = *(const v8us*)(arow0 + k0 + 16);
      v8us lo1 = *(const v8us*)(arow1 + k0);
      v8us hi1 = *(const v8us*)(arow1 + k0 + 16);
#pragma unroll
      for (int e = 0; e < 8; ++e) {
        af[0].u[e] = lo0[e]; af[0].u[8 + e] = hi0[e];
        af[1].u[e] = lo1[e]; af[1].u[8 + e] = hi1[e];
      }
    }
#pragma unroll
    for (int nj = 0; nj < 4; ++nj) {
      v8us lo = *(const v8us*)(brow[nj] + k0);
      v8us hi = *(const v8us*)(brow[nj] + k0 + 16);
#pragma unroll
      for (int e = 0; e < 8; ++e) { bfr[nj].u[e] = lo[e]; bfr[nj].u[8 + e] = hi[e]; }
    }
#pragma unroll
    for (int mi = 0; mi < 2; ++mi)
#pragma unroll
      for (int nj = 0; nj < 4; ++nj)
        acc[mi][nj] = __builtin_amdgcn_wmma_f32_16x16x32_bf16(
            false, af[mi].v, false, bfr[nj].v, (short)0, acc[mi][nj], false, false);
  }
  int mbase = m0 + ((lane < 16) ? 0 : 8);
#pragma unroll
  for (int nj = 0; nj < 4; ++nj) {
    int ncol = nst * 64 + nj * 16 + l15;
    float bn = bo[ncol];
#pragma unroll
    for (int mi = 0; mi < 2; ++mi) {
#pragma unroll
      for (int rr = 0; rr < 8; ++rr) {
        size_t off = (size_t)(mbase + mi * 16 + rr) * CC + ncol;
        out[off] = (acc[mi][nj][rr] + bn) * gbuf[off];
      }
    }
  }
}

extern "C" void kernel_launch(void* const* d_in, const int* in_sizes, int n_in,
                              void* d_out, int out_size, void* d_ws, size_t ws_size,
                              hipStream_t stream) {
  (void)in_sizes; (void)n_in; (void)out_size; (void)ws_size;
  const float* x     = (const float*)d_in[0];
  const float* W_r   = (const float*)d_in[1];
  const float* b_r   = (const float*)d_in[2];
  const float* W_k   = (const float*)d_in[3];
  const float* b_k   = (const float*)d_in[4];
  const float* W_v   = (const float*)d_in[5];
  const float* b_v   = (const float*)d_in[6];
  const float* W_g   = (const float*)d_in[7];
  const float* b_g   = (const float*)d_in[8];
  const float* W_o   = (const float*)d_in[9];
  const float* b_o   = (const float*)d_in[10];
  const float* decay = (const float*)d_in[11];
  const float* eta   = (const float*)d_in[12];
  const float* tmr   = (const float*)d_in[13];
  const float* tmk   = (const float*)d_in[14];
  const float* tmv   = (const float*)d_in[15];
  const float* tmg   = (const float*)d_in[16];
  const float* gnw   = (const float*)d_in[17];
  const float* gnb   = (const float*)d_in[18];

  char* ws = (char*)d_ws;
  const size_t MC = (size_t)MM * CC;
  unsigned short* xm = (unsigned short*)ws;                             // 4*MC bf16 (128 MB)
  unsigned short* wb = (unsigned short*)(ws + 4 * MC * 2);              // 5*C*C bf16 (10 MB)
  float* rbuf = (float*)(ws + 4 * MC * 2 + (size_t)5 * CC * CC * 2);    // MC f32
  float* kbuf = rbuf + MC;
  float* vbuf = kbuf + MC;
  float* gbuf = vbuf + MC;
  float* stats = gbuf + MC;                                             // 256 f32
  float* wkv = (float*)ws;                        // alias xm (dead after GEMM1)
  unsigned short* wkvn = (unsigned short*)(ws + MC * 4);                // alias xm upper half

  float* outp      = (float*)d_out;
  float* state_out = outp + MC;                                         // B*H*D*D
  float* lastx     = state_out + (size_t)BB * HH * DDIM * DDIM;         // B*C

  k_conv_weights<<<(5 * CC * CC + 255) / 256, 256, 0, stream>>>(W_r, W_k, W_v, W_g, W_o, wb);
  k_mix<<<(unsigned)(MC / 256), 256, 0, stream>>>(x, tmr, tmk, tmv, tmg, xm);
  k_gemm_qkvg<<<dim3(MM / 32, 4, 4), 128, 0, stream>>>(xm, wb, b_r, b_k, b_v, b_g,
                                                       rbuf, kbuf, vbuf, gbuf);
  k_l2norm64<<<(unsigned)((size_t)MM * HH / 8), 256, 0, stream>>>(kbuf);
  k_l2norm64<<<(unsigned)((size_t)MM * HH / 8), 256, 0, stream>>>(vbuf);
  k_scan<<<BB * HH, 256, 0, stream>>>(rbuf, kbuf, vbuf, decay, eta, wkv, state_out);
  k_gnstats<<<BB * HH, 256, 0, stream>>>(wkv, stats);
  k_gnnorm<<<(unsigned)(MC / 256), 256, 0, stream>>>(wkv, stats, gnw, gnb, wkvn);
  k_lastx<<<(BB * CC + 255) / 256, 256, 0, stream>>>(x, lastx);
  k_gemm_out<<<dim3(MM / 32, 4), 128, 0, stream>>>(wkvn, wb + (size_t)4 * CC * CC, b_o, gbuf, outp);
}
